// FdewetMPNN_23381801960016
// MI455X (gfx1250) — matrix operand order repaired and checked by reading, not compiled, not run on hardware
//
#include <hip/hip_runtime.h>
#include <math.h>

// ---------------- types for WMMA ----------------
typedef __bf16  bf16_t;
typedef bf16_t  v16bf __attribute__((ext_vector_type(16)));
typedef float   v8f   __attribute__((ext_vector_type(8)));

#define N_HID 24
#define EDIM  84
#define EPAD  96     // padded bf16 e-row width (16B-aligned rows)
#define NW    4      // waves per block in edge kernel (block = 128 threads)

// native bf16 convert (single v_cvt op on gfx1250, RNE)
__device__ __forceinline__ unsigned short f2bf(float f) {
  union { __bf16 h; unsigned short u; } c;
  c.h = (__bf16)f;
  return c.u;
}
__device__ __forceinline__ unsigned int f2bf2(float lo, float hi) {
  return (unsigned int)f2bf(lo) | ((unsigned int)f2bf(hi) << 16);
}

__device__ __forceinline__ float gelu_exact(float x) {
  return 0.5f * x * (1.0f + erff(x * 0.70710678118654752f));
}

union FragU { uint4 u[2]; v16bf v; };

__device__ __forceinline__ v16bf frag_ld2(const unsigned short* p0,
                                          const unsigned short* p1) {
  FragU f;
  f.u[0] = *(const uint4*)p0;
  f.u[1] = *(const uint4*)p1;
  return f.v;
}

// ---------------- weight prep: f32 [k][n] -> bf16 transposed padded [n][kpad] ----------------
__global__ void prep_weights_kernel(
    const float* __restrict__ pe_w1, const float* __restrict__ pe_w2,
    const float* __restrict__ pv_w1, const float* __restrict__ pv_w2,
    unsigned short* __restrict__ W1T, unsigned short* __restrict__ W2T,
    unsigned short* __restrict__ V1T, unsigned short* __restrict__ V2T)
{
  int idx = blockIdx.x * blockDim.x + threadIdx.x;
  const int n1 = 2 * 96 * 160;   // phi_e W1: 132x84 -> [96][160]
  const int n2 = 2 * 96 * 96;    // phi_e W2: 84x84  -> [96][96]
  const int n3 = 2 * 32 * 160;   // phi_v W1: 132x24 -> [32][160]
  const int n4 = 2 * 32 * 32;    // phi_v W2: 24x24  -> [32][32]
  if (idx < n1) {
    int l = idx / (96*160), rem = idx % (96*160);
    int n = rem / 160, k = rem % 160;
    float v = (n < 84 && k < 132) ? pe_w1[((size_t)l*132 + k)*84 + n] : 0.0f;
    W1T[idx] = f2bf(v);
    return;
  }
  idx -= n1;
  if (idx < n2) {
    int l = idx / (96*96), rem = idx % (96*96);
    int n = rem / 96, k = rem % 96;
    float v = (n < 84 && k < 84) ? pe_w2[((size_t)l*84 + k)*84 + n] : 0.0f;
    W2T[idx] = f2bf(v);
    return;
  }
  idx -= n2;
  if (idx < n3) {
    int l = idx / (32*160), rem = idx % (32*160);
    int n = rem / 160, k = rem % 160;
    float v = (n < 24 && k < 132) ? pv_w1[((size_t)l*132 + k)*24 + n] : 0.0f;
    V1T[idx] = f2bf(v);
    return;
  }
  idx -= n3;
  if (idx < n4) {
    int l = idx / (32*32), rem = idx % (32*32);
    int n = rem / 32, k = rem % 32;
    float v = (n < 24 && k < 24) ? pv_w2[((size_t)l*24 + k)*24 + n] : 0.0f;
    V2T[idx] = f2bf(v);
  }
}

// edge_attr f32 [E][84] -> bf16 mirror [E][96] (pad cols zero)
__global__ void econv_kernel(const float* __restrict__ ea,
                             unsigned short* __restrict__ ebf, long total) {
  long idx = (long)blockIdx.x * blockDim.x + threadIdx.x;
  if (idx >= total) return;
  long ei = idx / EPAD;
  int  k  = (int)(idx % EPAD);
  ebf[idx] = (k < EDIM) ? f2bf(ea[ei * EDIM + k]) : (unsigned short)0;
}

// ---------------- misc small kernels ----------------
__global__ void zero_kernel(float* __restrict__ p, int n) {
  int i = blockIdx.x * blockDim.x + threadIdx.x;
  if (i < n) p[i] = 0.0f;
}

__global__ void degree_kernel(const int* __restrict__ dstI,
                              float* __restrict__ cnt, int nE) {
  int i = blockIdx.x * blockDim.x + threadIdx.x;
  if (i < nE) atomicAdd(&cnt[dstI[i]], 1.0f);
}

// h = x @ node_w + node_b ; intrinsic head ; bf16 mirror of h
__global__ void node_input_kernel(
    const float* __restrict__ x, const float* __restrict__ nw,
    const float* __restrict__ nb, const float* __restrict__ iw1,
    const float* __restrict__ ib1, const float* __restrict__ iw2,
    const float* __restrict__ ib2,
    float* __restrict__ h, unsigned short* __restrict__ hbf,
    float* __restrict__ intr, int N)
{
  int n = blockIdx.x * blockDim.x + threadIdx.x;
  if (n >= N) return;
  float xr[32];
#pragma unroll
  for (int k = 0; k < 32; ++k) xr[k] = x[(size_t)n*32 + k];
  float hv[24];
#pragma unroll
  for (int j = 0; j < 24; ++j) {
    float s = nb[j];
#pragma unroll
    for (int k = 0; k < 32; ++k) s += xr[k] * nw[k*24 + j];
    hv[j] = s;
    h[(size_t)n*24 + j] = s;
    hbf[(size_t)n*24 + j] = f2bf(s);
  }
  float acc = ib2[0];
#pragma unroll
  for (int j = 0; j < 20; ++j) {
    float s = ib1[j];
#pragma unroll
    for (int k = 0; k < 24; ++k) s += hv[k] * iw1[k*20 + j];
    acc += gelu_exact(s) * iw2[j];
  }
  intr[n] = acc;
}

// h = LN(h + agg / max(counts,1)) ; refresh bf16 mirror
__global__ void node_update_kernel(
    float* __restrict__ h, unsigned short* __restrict__ hbf,
    const float* __restrict__ agg, const float* __restrict__ cnt,
    const float* __restrict__ g, const float* __restrict__ b, int N)
{
  int n = blockIdx.x * blockDim.x + threadIdx.x;
  if (n >= N) return;
  float c = cnt[n]; if (c < 1.0f) c = 1.0f;
  float rc = 1.0f / c;
  float v[24]; float s1 = 0.f, s2 = 0.f;
#pragma unroll
  for (int j = 0; j < 24; ++j) {
    float t = h[(size_t)n*24 + j] + agg[(size_t)n*24 + j] * rc;
    v[j] = t; s1 += t; s2 += t*t;
  }
  const float mean = s1 * (1.0f/24.0f);
  const float var  = s2 * (1.0f/24.0f) - mean*mean;
  const float rst  = rsqrtf(var + 1e-5f);
#pragma unroll
  for (int j = 0; j < 24; ++j) {
    float t = (v[j] - mean) * rst * g[j] + b[j];
    h[(size_t)n*24 + j] = t;
    hbf[(size_t)n*24 + j] = f2bf(t);
  }
}

// out = mu + intrinsic + context(h)
__global__ void final_kernel(
    const float* __restrict__ h, const float* __restrict__ intr,
    const float* __restrict__ cw1, const float* __restrict__ cb1,
    const float* __restrict__ cw2, const float* __restrict__ cb2,
    const float* __restrict__ mu, float* __restrict__ out, int N)
{
  int n = blockIdx.x * blockDim.x + threadIdx.x;
  if (n >= N) return;
  float hv[24];
#pragma unroll
  for (int k = 0; k < 24; ++k) hv[k] = h[(size_t)n*24 + k];
  float acc = cb2[0];
#pragma unroll
  for (int j = 0; j < 20; ++j) {
    float s = cb1[j];
#pragma unroll
    for (int k = 0; k < 24; ++k) s += hv[k] * cw1[k*20 + j];
    acc += gelu_exact(s) * cw2[j];
  }
  out[n] = mu[0] + intr[n] + acc;
}

// ---------------- fused per-layer edge kernel (WMMA core) ----------------
// One wave32 owns a 16-edge tile: gather cat=[hs|hd|e] (bf16, LDS, vectorized),
// phi_e MLP (WMMA bf16->f32) + residual + LN (f32, lane-pair parallel),
// write e (f32 + bf16 mirror) back, phi_v MLP (WMMA), atomic scatter into agg.
__global__ void __launch_bounds__(128) edge_layer_kernel(
    const float* __restrict__ ein, float* __restrict__ eout,
    unsigned short* __restrict__ ebf,
    const unsigned short* __restrict__ hbf,
    const int* __restrict__ srcI, const int* __restrict__ dstI,
    const unsigned short* __restrict__ W1T, const unsigned short* __restrict__ W2T,
    const unsigned short* __restrict__ V1T, const unsigned short* __restrict__ V2T,
    const float* __restrict__ eb1, const float* __restrict__ eb2,
    const float* __restrict__ lng, const float* __restrict__ lnb,
    const float* __restrict__ vb1, const float* __restrict__ vb2,
    float* __restrict__ agg, int nE)
{
  __shared__ __align__(16) unsigned short catS[NW][16][160]; // cat tile, K padded to 160
  __shared__ __align__(16) unsigned short tmpS[NW][16][96];  // GELU intermediates, K padded
  __shared__ float lnr[NW][16][88];                          // f32 row buffer for residual+LN

  const int wave = threadIdx.x >> 5;
  const int lane = threadIdx.x & 31;
  const int tile = blockIdx.x * NW + wave;
  const int ebase = tile * 16;
  if (ebase >= nE) return;                      // wave-uniform exit, no barriers used

  // lane-pair row assignment for gather / LN
  const int r  = lane >> 1;          // edge row within tile
  const int hf = lane & 1;           // half (0/1)
  int eic = ebase + r; if (eic > nE - 1) eic = nE - 1;   // clamped; stores guarded later

  // load gather indices early + prefetch gathered rows (global_prefetch_b8)
  const int nid = (hf == 0) ? srcI[eic] : dstI[eic];
  __builtin_prefetch(hbf + (size_t)nid * 24, 0, 3);
  __builtin_prefetch(ebf + (size_t)eic * EPAD + hf * 48, 0, 3);

  // zero K-pad regions (wave-private LDS; overlaps prefetch latency)
  for (int i = lane; i < 16*28; i += 32) catS[wave][i/28][132 + (i%28)] = 0;
  for (int i = lane; i < 16*12; i += 32) tmpS[wave][i/12][84 + (i%12)] = 0;

  // ---- gather: all-vector loads. cat row = [hs(24) | hd(24) | e(84, bf16 mirror)] ----
  {
    const uint4* hp = (const uint4*)(hbf + (size_t)nid * 24);   // 48B, 16B-aligned
    if (hf == 0) {
      uint4* cp = (uint4*)&catS[wave][r][0];
      cp[0] = hp[0]; cp[1] = hp[1]; cp[2] = hp[2];
      const uint4* ep = (const uint4*)(ebf + (size_t)eic * EPAD);
      uint4* cq = (uint4*)&catS[wave][r][48];
#pragma unroll
      for (int q = 0; q < 6; ++q) cq[q] = ep[q];                // e cols 0..47
    } else {
      uint4* cp = (uint4*)&catS[wave][r][24];
      cp[0] = hp[0]; cp[1] = hp[1]; cp[2] = hp[2];
      const uint4* ep = (const uint4*)(ebf + (size_t)eic * EPAD + 48);
      uint4* cq = (uint4*)&catS[wave][r][96];
#pragma unroll
      for (int q = 0; q < 5; ++q) cq[q] = ep[q];                // e cols 48..87 (84..87 = 0 pad)
    }
  }

  // per-lane fragment geometry (CDNA5 16-bit A / B / f32 C layouts)
  const int m     = lane & 15;           // A row
  const int koffA = (lane >> 4) << 3;    // A K sub-offset: 0 / 8
  const int khB   = (lane >> 4) << 4;    // unused (B fragments use koffA, same K order as A)
  const int mrow  = (lane >> 4) << 3;    // C row offset: 0 / 8
  const int nloc  = lane & 15;           // C/B column within N-tile

  v16bf afr[5];

  // ---- phi_e GEMM1: cat[16x132] @ W1[132x84] -> GELU -> tmpS ----
#pragma unroll
  for (int kt = 0; kt < 5; ++kt) {
    const unsigned short* pa = &catS[wave][m][kt*32 + koffA];
    afr[kt] = frag_ld2(pa, pa + 16);
  }
#pragma unroll
  for (int nt = 0; nt < 6; ++nt) {
    v8f acc = {};
#pragma unroll
    for (int kt = 0; kt < 5; ++kt) {
      const unsigned short* pb = W1T + (size_t)(nt*16 + nloc)*160 + kt*32 + koffA;
      v16bf bfr = frag_ld2(pb, pb + 16);
      acc = __builtin_amdgcn_wmma_f32_16x16x32_bf16(false, afr[kt], false, bfr,
                                                    (short)0, acc, false, false);
    }
    const int n = nt*16 + nloc;
    if (n < EDIM) {
      const float b = eb1[n];
#pragma unroll
      for (int rr = 0; rr < 8; ++rr)
        tmpS[wave][mrow + rr][n] = f2bf(gelu_exact(acc[rr] + b));
    }
  }

  // ---- phi_e GEMM2: tmpS[16x84] @ W2[84x84] -> lnr (f32 msg) ----
  {
    v16bf a2[3];
#pragma unroll
    for (int kt = 0; kt < 3; ++kt) {
      const unsigned short* pa = &tmpS[wave][m][kt*32 + koffA];
      a2[kt] = frag_ld2(pa, pa + 16);
    }
#pragma unroll
    for (int nt = 0; nt < 6; ++nt) {
      v8f acc = {};
#pragma unroll
      for (int kt = 0; kt < 3; ++kt) {
        const unsigned short* pb = W2T + (size_t)(nt*16 + nloc)*96 + kt*32 + koffA;
        v16bf bfr = frag_ld2(pb, pb + 16);
        acc = __builtin_amdgcn_wmma_f32_16x16x32_bf16(false, a2[kt], false, bfr,
                                                      (short)0, acc, false, false);
      }
      const int n = nt*16 + nloc;
      if (n < EDIM) {
        const float b = eb2[n];
#pragma unroll
        for (int rr = 0; rr < 8; ++rr)
          lnr[wave][mrow + rr][n] = acc[rr] + b;
      }
    }
  }

  // ---- residual + LayerNorm per edge row; lane-pair parallel (42 cols each) ----
  {
    const int ei = ebase + r;
    if (ei < nE) {
      const float* ep = ein + (size_t)ei * EDIM + hf * 42;
      float vals[42];
      float s1 = 0.f, s2 = 0.f;
#pragma unroll
      for (int j = 0; j < 42; ++j) {
        float v = ep[j] + lnr[wave][r][hf*42 + j];
        vals[j] = v; s1 += v; s2 += v*v;
      }
      s1 += __shfl_xor(s1, 1, 32);            // combine halves of the row
      s2 += __shfl_xor(s2, 1, 32);
      const float mean = s1 * (1.0f/EDIM);
      const float var  = s2 * (1.0f/EDIM) - mean*mean;
      const float rst  = rsqrtf(var + 1e-5f);
      float*          eo  = eout + (size_t)ei * EDIM + hf*42;
      unsigned short* eb  = ebf  + (size_t)ei * EPAD + hf*42;
      const float*    gg  = lng + hf*42;
      const float*    bb  = lnb + hf*42;
#pragma unroll
      for (int j = 0; j < 42; j += 2) {
        float t0 = (vals[j  ] - mean) * rst * gg[j  ] + bb[j  ];
        float t1 = (vals[j+1] - mean) * rst * gg[j+1] + bb[j+1];
        *(float2*)(eo + j) = make_float2(t0, t1);               // f32 state
        unsigned int pk = f2bf2(t0, t1);
        *(unsigned int*)(eb + j) = pk;                          // bf16 mirror
        *(unsigned int*)(&catS[wave][r][48 + hf*42 + j]) = pk;  // refresh cat tile
      }
    }
  }

  // ---- phi_v GEMM1: cat[16x132] @ V1[132x24] -> GELU -> tmpS[16x32] (zero-padded) ----
#pragma unroll
  for (int kt = 0; kt < 5; ++kt) {
    const unsigned short* pa = &catS[wave][m][kt*32 + koffA];
    afr[kt] = frag_ld2(pa, pa + 16);
  }
#pragma unroll
  for (int nt = 0; nt < 2; ++nt) {
    v8f acc = {};
#pragma unroll
    for (int kt = 0; kt < 5; ++kt) {
      const unsigned short* pb = V1T + (size_t)(nt*16 + nloc)*160 + kt*32 + koffA;
      v16bf bfr = frag_ld2(pb, pb + 16);
      acc = __builtin_amdgcn_wmma_f32_16x16x32_bf16(false, afr[kt], false, bfr,
                                                    (short)0, acc, false, false);
    }
    const int n = nt*16 + nloc;
    const float b = (n < N_HID) ? vb1[n] : 0.f;
#pragma unroll
    for (int rr = 0; rr < 8; ++rr) {
      float v = (n < N_HID) ? gelu_exact(acc[rr] + b) : 0.f;
      tmpS[wave][mrow + rr][n] = f2bf(v);
    }
  }

  // ---- phi_v GEMM2: tmpS[16x24] @ V2[24x24] -> atomic scatter-add into agg ----
  {
    const unsigned short* pa = &tmpS[wave][m][koffA];
    v16bf a3 = frag_ld2(pa, pa + 16);
#pragma unroll
    for (int nt = 0; nt < 2; ++nt) {
      const unsigned short* pb = V2T + (size_t)(nt*16 + nloc)*32 + koffA;
      v16bf bfr = frag_ld2(pb, pb + 16);
      v8f acc = {};
      acc = __builtin_amdgcn_wmma_f32_16x16x32_bf16(false, a3, false, bfr,
                                                    (short)0, acc, false, false);
      const int n = nt*16 + nloc;
      if (n < N_HID) {
        const float b = vb2[n];
#pragma unroll
        for (int rr = 0; rr < 8; ++rr) {
          const int ei2 = ebase + mrow + rr;
          if (ei2 < nE) {
            const int d = dstI[ei2];
            atomicAdd(&agg[(size_t)d * N_HID + n], acc[rr] + b);
          }
        }
      }
    }
  }
}

// ---------------- host launcher ----------------
extern "C" void kernel_launch(void* const* d_in, const int* in_sizes, int n_in,
                              void* d_out, int out_size, void* d_ws, size_t ws_size,
                              hipStream_t stream)
{
  (void)n_in; (void)out_size; (void)ws_size;
  const float* x      = (const float*)d_in[0];
  const float* ea     = (const float*)d_in[1];
  const float* node_w = (const float*)d_in[2];
  const float* node_b = (const float*)d_in[3];
  const float* ih_w1  = (const float*)d_in[4];
  const float* ih_b1  = (const float*)d_in[5];
  const float* ih_w2  = (const float*)d_in[6];
  const float* ih_b2  = (const float*)d_in[7];
  const float* pe_w1  = (const float*)d_in[8];
  const float* pe_b1  = (const float*)d_in[9];
  const float* pe_w2  = (const float*)d_in[10];
  const float* pe_b2  = (const float*)d_in[11];
  const float* pv_w1  = (const float*)d_in[12];
  const float* pv_b1  = (const float*)d_in[13];
  const float* pv_w2  = (const float*)d_in[14];
  const float* pv_b2  = (const float*)d_in[15];
  const float* ne_g   = (const float*)d_in[16];
  const float* ne_b   = (const float*)d_in[17];
  const float* nv_g   = (const float*)d_in[18];
  const float* nv_b   = (const float*)d_in[19];
  const float* ch_w1  = (const float*)d_in[20];
  const float* ch_b1  = (const float*)d_in[21];
  const float* ch_w2  = (const float*)d_in[22];
  const float* ch_b2  = (const float*)d_in[23];
  const float* mu     = (const float*)d_in[24];
  const int*   eidx   = (const int*)d_in[25];

  const int N = in_sizes[0] / 32;
  const int E = in_sizes[1] / EDIM;
  const int* srcI = eidx;
  const int* dstI = eidx + E;

  // carve workspace (256B-aligned regions)
  char* p = (char*)d_ws;
  auto take = [&](size_t bytes) -> char* {
    char* r = p; p += (bytes + 255) & ~(size_t)255; return r;
  };
  unsigned short* W1T = (unsigned short*)take((size_t)2*96*160*2);
  unsigned short* W2T = (unsigned short*)take((size_t)2*96*96*2);
  unsigned short* V1T = (unsigned short*)take((size_t)2*32*160*2);
  unsigned short* V2T = (unsigned short*)take((size_t)2*32*32*2);
  float*          h   = (float*)take((size_t)N*24*4);
  unsigned short* hbf = (unsigned short*)take((size_t)N*24*2);
  float*          e   = (float*)take((size_t)E*EDIM*4);
  unsigned short* ebf = (unsigned short*)take((size_t)E*EPAD*2);
  float*          agg = (float*)take((size_t)N*24*4);
  float*          cnt = (float*)take((size_t)N*4);
  float*          intr= (float*)take((size_t)N*4);
  float* out = (float*)d_out;

  prep_weights_kernel<<<240, 256, 0, stream>>>(pe_w1, pe_w2, pv_w1, pv_w2,
                                               W1T, W2T, V1T, V2T);
  {
    long total = (long)E * EPAD;
    long blocks = (total + 255) / 256;
    econv_kernel<<<(int)blocks, 256, 0, stream>>>(ea, ebf, total);
  }
  zero_kernel<<<(N + 255)/256, 256, 0, stream>>>(cnt, N);
  degree_kernel<<<(E + 255)/256, 256, 0, stream>>>(dstI, cnt, E);
  node_input_kernel<<<(N + 255)/256, 256, 0, stream>>>(
      x, node_w, node_b, ih_w1, ih_b1, ih_w2, ih_b2, h, hbf, intr, N);

  const int tiles = (E + 15) / 16;
  const int eblocks = (tiles + NW - 1) / NW;
  for (int l = 0; l < 2; ++l) {
    zero_kernel<<<(N*24 + 255)/256, 256, 0, stream>>>(agg, N*24);
    edge_layer_kernel<<<eblocks, 128, 0, stream>>>(
        (l == 0) ? ea : e, e, ebf, hbf, srcI, dstI,
        W1T + (size_t)l*96*160, W2T + (size_t)l*96*96,
        V1T + (size_t)l*32*160, V2T + (size_t)l*32*32,
        pe_b1 + l*EDIM, pe_b2 + l*EDIM, ne_g + l*EDIM, ne_b + l*EDIM,
        pv_b1 + l*N_HID, pv_b2 + l*N_HID, agg, E);
    node_update_kernel<<<(N + 255)/256, 256, 0, stream>>>(
        h, hbf, agg, cnt, nv_g + l*N_HID, nv_b + l*N_HID, N);
  }
  final_kernel<<<(N + 255)/256, 256, 0, stream>>>(
      h, intr, ch_w1, ch_b1, ch_w2, ch_b2, mu, out, N);
}
